// Decoder_31267361915186
// MI455X (gfx1250) — compile-verified, matrix operand order checked by reading, not run on hardware
//
#include <hip/hip_runtime.h>
#include <hip/hip_bf16.h>

// ---------------------------------------------------------------------------
// Attention LSTM decoder for MI455X (gfx1250, wave32, WMMA + TDM).
// All GEMMs run through V_WMMA_F32_16X16X32_BF16 (bf16 inputs, f32 accum).
// Weights are cast to bf16 once per launch into d_ws; the whole bf16 weight
// set (~115MB) is L2-resident (192MB), so the 48 sequential steps hit L2.
// For the per-step GEMMs (M=32) the A operand is staged into LDS with the
// Tensor Data Mover (tensor_load_to_lds) once per workgroup, and read back
// with true ds_load_b128 (explicit addrspace(3) pointers) so the VMEM path
// is dedicated to the streaming weight operand.
// ---------------------------------------------------------------------------

typedef __bf16 bf16_t;
typedef __attribute__((ext_vector_type(16))) __bf16 v16bf;
typedef __attribute__((ext_vector_type(8)))  __bf16 v8bf;
typedef __attribute__((ext_vector_type(8)))  float   v8f;
typedef __attribute__((ext_vector_type(4)))  unsigned int u32x4;
typedef __attribute__((ext_vector_type(8)))  int          i32x8;

#define NB   32     // batch
#define NS   64     // source length
#define NE   512    // embed dim
#define ND   1024   // decoder hidden
#define NT   48     // target length
#define NV   32000  // vocab
#define TWOE 1024   // 2 * encoder hidden
#define NG   4096   // 4*ND gates

// ---------------------------------------------------------------------------
// TDM: load a rows x cols (bf16) 2D tile from global (row stride
// src_stride_elems) into LDS at byte offset lds_off. D# per ISA 8.3/8.4:
//   group0: count=1 | lds_addr | global_addr[56:0] | type=2
//   group1: data_size=1(2B), tensor_dim0=cols, tensor_dim1=rows,
//           tile_dim0=cols, tile_dim1=rows, tensor_dim0_stride=stride
// 2D tensor -> VADDR2/VADDR3 are NULL (2-operand asm form). Descriptors are
// uniform, so they live in SGPRs. Tracked by TENSORcnt.
// ---------------------------------------------------------------------------
__device__ __forceinline__ void tdm_load_2d(const bf16_t* gsrc, unsigned lds_off,
                                            int rows, int cols,
                                            int src_stride_elems) {
  unsigned long long ga = (unsigned long long)(uintptr_t)gsrc;
  u32x4 g0;
  g0[0] = 1u;                                        // count=1 (valid user D#)
  g0[1] = lds_off;                                   // lds_addr (bytes)
  g0[2] = (unsigned)(ga & 0xFFFFFFFFull);            // global_addr[31:0]
  g0[3] = (unsigned)((ga >> 32) & 0x1FFFFFFull)      // global_addr[56:32]
        | (2u << 30);                                // type = 2 ("image")
  i32x8 g1;
  g1[0] = (1 << 16);                                 // data_size=1 (2 bytes)
  g1[1] = (cols & 0xFFFF) << 16;                     // tensor_dim0[15:0]
  g1[2] = ((cols >> 16) & 0xFFFF)                    // tensor_dim0[31:16]
        | ((rows & 0xFFFF) << 16);                   // tensor_dim1[15:0]
  g1[3] = ((rows >> 16) & 0xFFFF)                    // tensor_dim1[31:16]
        | ((cols & 0xFFFF) << 16);                   // tile_dim0 = cols
  g1[4] = (rows & 0xFFFF);                           // tile_dim1 = rows
  g1[5] = src_stride_elems;                          // dim0_stride[31:0]
  g1[6] = 0;                                         // dim0_stride[47:32]
  g1[7] = 0;
  asm volatile("tensor_load_to_lds %0, %1" :: "s"(g0), "s"(g1) : "memory");
}

// ---------------------------------------------------------------------------
// WMMA fragment helpers.
// 16-bit A-matrix 16x32 layout (ISA 7.12.2): lane L (L<16) holds row M=L,
// K = {kb+0..7, kb+16..23}; lane L>=16 holds row M=L-16, K = {kb+8..15,
// kb+24..31}. We assume the B operand (K x N) mirrors this with lane = column
// N, so weight matrices stored row-major as [N,K] (i.e. x @ W^T) load with the
// identical per-lane indexing.
// ---------------------------------------------------------------------------
__device__ __forceinline__ v16bf load_frag_row(const bf16_t* __restrict__ p) {
  // p = row_base + kbase + ((lane&16) ? 8 : 0); two 16B chunks 16 elems apart
  v8bf a = *(const v8bf*)(p);
  v8bf b = *(const v8bf*)(p + 16);
  v16bf r;
#pragma unroll
  for (int i = 0; i < 8; ++i) { r[i] = a[i]; r[i + 8] = b[i]; }
  return r;
}

// Same fragment load but from LDS via an explicit addrspace(3) pointer
// (32-bit LDS byte offset) -> guaranteed ds_load_b128.
typedef __attribute__((address_space(3))) v8bf lds_v8bf;
__device__ __forceinline__ v16bf load_frag_row_lds(unsigned byte_off) {
  v8bf a = *(const lds_v8bf*)(byte_off);
  v8bf b = *(const lds_v8bf*)(byte_off + 32u);   // +16 bf16 elements
  v16bf r;
#pragma unroll
  for (int i = 0; i < 8; ++i) { r[i] = a[i]; r[i + 8] = b[i]; }
  return r;
}

// ---------------------------------------------------------------------------
// Generic dual-input WMMA GEMM:
//   C[M,N] = act( A0[M,K0] @ W0[N,K0]^T + A1[M,K1] @ W1[N,K1]^T + b0 + b1 )
// One wave per 16x16 tile. All dims multiples of 16 (K multiples of 32).
// STAGE=1: TDM-stage full A0 (and A1) [M x K] into LDS once per block;
// fragment A reads then come from LDS (ds_load_b128) while VMEM streams W.
// ---------------------------------------------------------------------------
template <int STAGE>
__global__ void wmma_gemm_kernel(
    const bf16_t* __restrict__ A0, int lda0, int K0, const bf16_t* __restrict__ W0,
    const bf16_t* __restrict__ A1, int lda1, int K1, const bf16_t* __restrict__ W1,
    const float* __restrict__ bias0, const float* __restrict__ bias1,
    float* __restrict__ C, int ldc,
    bf16_t* __restrict__ Cbf, int ldcb,
    int M, int N, int apply_tanh)
{
  extern __shared__ bf16_t lds_a[];
  const int lane = threadIdx.x & 31;
  const int wave = blockIdx.x * (blockDim.x >> 5) + (threadIdx.x >> 5);
  const int tiles_m = M >> 4;
  const int tiles_n = N >> 4;

  unsigned abase = 0;
  if (STAGE) {
    abase = (unsigned)(uintptr_t)lds_a;   // low 32 bits = LDS byte offset
    if ((threadIdx.x >> 5) == 0) {   // one wave issues the TDM ops (EXEC ignored)
      tdm_load_2d(A0, abase, M, K0, lda0);
      if (W1) tdm_load_2d(A1, abase + (unsigned)(M * K0 * 2), M, K1, lda1);
      __builtin_amdgcn_s_wait_tensorcnt((short)0);
    }
    __syncthreads();
  }
  if (wave >= tiles_m * tiles_n) return;       // uniform per wave

  const int mt   = wave % tiles_m;
  const int nt   = wave / tiles_m;
  const int r    = lane & 15;
  const int koff = (lane & 16) ? 8 : 0;

  v8f acc = {};
  const bf16_t* w0row = W0 + (long long)(nt * 16 + r) * K0 + koff;

  if (STAGE) {
    const unsigned a0off = abase + (unsigned)(((mt * 16 + r) * K0 + koff) * 2);
    for (int k = 0; k < K0; k += 32) {
      if (k + 128 < K0) __builtin_prefetch(w0row + k + 128, 0, 3);
      v16bf a = load_frag_row_lds(a0off + (unsigned)(2 * k));
      v16bf b = load_frag_row(w0row + k);
      acc = __builtin_amdgcn_wmma_f32_16x16x32_bf16(
          false, a, false, b, (short)0, acc, false, false);
    }
    if (W1) {
      const unsigned a1off = abase + (unsigned)(M * K0 * 2)
                           + (unsigned)(((mt * 16 + r) * K1 + koff) * 2);
      const bf16_t* w1row = W1 + (long long)(nt * 16 + r) * K1 + koff;
      for (int k = 0; k < K1; k += 32) {
        if (k + 128 < K1) __builtin_prefetch(w1row + k + 128, 0, 3);
        v16bf a = load_frag_row_lds(a1off + (unsigned)(2 * k));
        v16bf b = load_frag_row(w1row + k);
        acc = __builtin_amdgcn_wmma_f32_16x16x32_bf16(
            false, a, false, b, (short)0, acc, false, false);
      }
    }
  } else {
    const bf16_t* a0row = A0 + (long long)(mt * 16 + r) * lda0 + koff;
    for (int k = 0; k < K0; k += 32) {
      if (k + 128 < K0) __builtin_prefetch(w0row + k + 128, 0, 3);
      v16bf a = load_frag_row(a0row + k);
      v16bf b = load_frag_row(w0row + k);
      acc = __builtin_amdgcn_wmma_f32_16x16x32_bf16(
          false, a, false, b, (short)0, acc, false, false);
    }
    if (W1) {
      const bf16_t* a1row = A1 + (long long)(mt * 16 + r) * lda1 + koff;
      const bf16_t* w1row = W1 + (long long)(nt * 16 + r) * K1 + koff;
      for (int k = 0; k < K1; k += 32) {
        if (k + 128 < K1) __builtin_prefetch(w1row + k + 128, 0, 3);
        v16bf a = load_frag_row(a1row + k);
        v16bf b = load_frag_row(w1row + k);
        acc = __builtin_amdgcn_wmma_f32_16x16x32_bf16(
            false, a, false, b, (short)0, acc, false, false);
      }
    }
  }

  // C/D layout (ISA 7.12.2): VGPR i -> M = mt*16 + i + (lane>=16 ? 8 : 0),
  // N = nt*16 + (lane&15).
  const int n = nt * 16 + r;
  float bv = 0.f;
  if (bias0) bv += bias0[n];
  if (bias1) bv += bias1[n];
  const int mbase = mt * 16 + ((lane & 16) ? 8 : 0);
#pragma unroll
  for (int i = 0; i < 8; ++i) {
    float v = acc[i] + bv;
    if (apply_tanh) v = tanhf(v);
    if (C)   C  [(long long)(mbase + i) * ldc  + n] = v;
    if (Cbf) Cbf[(long long)(mbase + i) * ldcb + n] = (bf16_t)v;
  }
}

// ---------------------------------------------------------------------------
// f32 -> bf16 cast (grid stride)
// ---------------------------------------------------------------------------
__global__ void cast_bf16_kernel(const float* __restrict__ in,
                                 bf16_t* __restrict__ out, long long n) {
  long long i      = (long long)blockIdx.x * blockDim.x + threadIdx.x;
  long long stride = (long long)gridDim.x * blockDim.x;
  for (; i < n; i += stride) out[i] = (bf16_t)in[i];
}

// ---------------------------------------------------------------------------
// Initialize recurrent state: h=h0, c=c0, h_bf, and zero the prev-context
// half of the x buffer (context at t=0 is zeros).
// ---------------------------------------------------------------------------
__global__ void init_state_kernel(const float* __restrict__ h0,
                                  const float* __restrict__ c0,
                                  float* __restrict__ h, float* __restrict__ c,
                                  bf16_t* __restrict__ h_bf,
                                  bf16_t* __restrict__ x_bf) {
  int i = blockIdx.x * blockDim.x + threadIdx.x;
  if (i >= NB * ND) return;
  float hv = h0[i];
  h[i] = hv; c[i] = c0[i]; h_bf[i] = (bf16_t)hv;
  int b = i / ND, d = i % ND;
  x_bf[b * (NE + ND) + NE + d] = (bf16_t)0.f;
}

// ---------------------------------------------------------------------------
// Gather word embeddings for step t into x[:, 0:NE] (bf16).
// ---------------------------------------------------------------------------
__global__ void gather_emb_kernel(const float* __restrict__ emb,
                                  const int* __restrict__ tgt,
                                  bf16_t* __restrict__ x_bf, int t) {
  int i = blockIdx.x * blockDim.x + threadIdx.x;
  if (i >= NB * NE) return;
  int b = i / NE, e = i % NE;
  int tok = tgt[t * NB + b];
  x_bf[b * (NE + ND) + e] = (bf16_t)emb[(long long)tok * NE + e];
}

// ---------------------------------------------------------------------------
// LSTM pointwise update from gates [NB, 4*ND] (PyTorch order i,f,g,o).
// ---------------------------------------------------------------------------
__global__ void lstm_update_kernel(const float* __restrict__ g,
                                   float* __restrict__ h, float* __restrict__ c,
                                   bf16_t* __restrict__ h_bf) {
  int i = blockIdx.x * blockDim.x + threadIdx.x;
  if (i >= NB * ND) return;
  int b = i / ND, d = i % ND;
  const float* gb = g + (long long)b * NG;
  float ig = gb[d], fg = gb[ND + d], gg = gb[2 * ND + d], og = gb[3 * ND + d];
  float si = 1.f / (1.f + __expf(-ig));
  float sf = 1.f / (1.f + __expf(-fg));
  float so = 1.f / (1.f + __expf(-og));
  float cn = sf * c[i] + si * tanhf(gg);
  float hn = so * tanhf(cn);
  c[i] = cn; h[i] = hn; h_bf[i] = (bf16_t)hn;
}

// ---------------------------------------------------------------------------
// Attention for one step. One block per batch row b, 64 threads (one per s).
// logits[s] = <S_att[s*B+b,:], h[b,:]>; softmax; context = sum_s alpha*enc;
// writes hc_bf[b] = bf16([h, context])  (input to the combine GEMM).
// ---------------------------------------------------------------------------
__global__ void attention_kernel(const float* __restrict__ S_att, // [S*B, ND]
                                 const float* __restrict__ enc,   // [S,B,TWOE]
                                 const float* __restrict__ h,     // [B, ND]
                                 bf16_t* __restrict__ hc_bf) {    // [B, ND+TWOE]
  __shared__ float sl[NS];
  const int b = blockIdx.x;
  const int s = threadIdx.x;                 // 0..63
  const float* hb  = h + (long long)b * ND;
  const float* row = S_att + (long long)(s * NB + b) * ND;
  float acc = 0.f;
  for (int d = 0; d < ND; ++d) acc += row[d] * hb[d];
  sl[s] = acc;
  __syncthreads();
  float m = -INFINITY;
  for (int j = 0; j < NS; ++j) m = fmaxf(m, sl[j]);
  float e = __expf(sl[s] - m);
  __syncthreads();
  sl[s] = e;
  __syncthreads();
  float sum = 0.f;
  for (int j = 0; j < NS; ++j) sum += sl[j];
  float inv = 1.f / sum;
  for (int ec = 0; ec < TWOE / NS; ++ec) {
    int e2 = s * (TWOE / NS) + ec;
    float cv = 0.f;
    for (int j = 0; j < NS; ++j)
      cv += sl[j] * enc[((long long)j * NB + b) * TWOE + e2];
    hc_bf[(long long)b * (ND + TWOE) + ND + e2] = (bf16_t)(cv * inv);
  }
  for (int dc = 0; dc < ND / NS; ++dc) {
    int d = s * (ND / NS) + dc;
    hc_bf[(long long)b * (ND + TWOE) + d] = (bf16_t)hb[d];
  }
}

// ---------------------------------------------------------------------------
// Host side
// ---------------------------------------------------------------------------
static inline void launch_gemm(hipStream_t s,
    const bf16_t* A0, int lda0, int K0, const bf16_t* W0,
    const bf16_t* A1, int lda1, int K1, const bf16_t* W1,
    const float* b0, const float* b1,
    float* C, int ldc, bf16_t* Cbf, int ldcb,
    int M, int N, int tanh_flag, int stage_a) {
  int tiles  = (M / 16) * (N / 16);
  int wpb    = 8;                       // 8 waves (256 threads) per block
  int blocks = (tiles + wpb - 1) / wpb;
  if (stage_a) {
    size_t shmem = (size_t)2 * ((size_t)M * K0 + (W1 ? (size_t)M * K1 : 0));
    wmma_gemm_kernel<1><<<blocks, wpb * 32, shmem, s>>>(
        A0, lda0, K0, W0, A1, lda1, K1, W1, b0, b1,
        C, ldc, Cbf, ldcb, M, N, tanh_flag);
  } else {
    wmma_gemm_kernel<0><<<blocks, wpb * 32, 0, s>>>(
        A0, lda0, K0, W0, A1, lda1, K1, W1, b0, b1,
        C, ldc, Cbf, ldcb, M, N, tanh_flag);
  }
}

static inline void launch_cast(hipStream_t s, const float* in, bf16_t* out,
                               long long n) {
  long long blocks = (n + 255) / 256;
  if (blocks > 65535 * 8) blocks = 65535 * 8;
  cast_bf16_kernel<<<(int)blocks, 256, 0, s>>>(in, out, n);
}

extern "C" void kernel_launch(void* const* d_in, const int* in_sizes, int n_in,
                              void* d_out, int out_size, void* d_ws, size_t ws_size,
                              hipStream_t stream) {
  const float* src_enc  = (const float*)d_in[0];   // [S,B,TWOE]
  const float* h0       = (const float*)d_in[1];
  const float* c0       = (const float*)d_in[2];
  const float* emb      = (const float*)d_in[3];   // [V,NE]
  const float* W_proj   = (const float*)d_in[4];   // [ND,TWOE]
  const float* W_comb   = (const float*)d_in[5];   // [ND, TWOE+ND]
  const float* W_out    = (const float*)d_in[6];   // [V, ND]
  const float* W_ih0    = (const float*)d_in[7];   // [NG, NE+ND]
  const float* W_hh0    = (const float*)d_in[8];   // [NG, ND]
  const float* b_ih0    = (const float*)d_in[9];
  const float* b_hh0    = (const float*)d_in[10];
  const float* W_ih1    = (const float*)d_in[11];  // [NG, ND]
  const float* W_hh1    = (const float*)d_in[12];  // [NG, ND]
  const float* b_ih1    = (const float*)d_in[13];
  const float* b_hh1    = (const float*)d_in[14];
  const int*   tgt      = (const int*)d_in[15];    // [T,B]
  float*       out      = (float*)d_out;           // [T,B,V]

  // ---- workspace carve-up (256B aligned) ----
  char* base = (char*)d_ws;
  size_t off = 0;
  auto carve = [&](size_t bytes) -> char* {
    off = (off + 255) & ~(size_t)255;
    char* p = base + off;
    off += bytes;
    return p;
  };
  bf16_t* Wih0_bf = (bf16_t*)carve((size_t)NG * (NE + ND) * 2);
  bf16_t* Whh0_bf = (bf16_t*)carve((size_t)NG * ND * 2);
  bf16_t* Wih1_bf = (bf16_t*)carve((size_t)NG * ND * 2);
  bf16_t* Whh1_bf = (bf16_t*)carve((size_t)NG * ND * 2);
  bf16_t* Wcmb_bf = (bf16_t*)carve((size_t)ND * (TWOE + ND) * 2);
  bf16_t* Wout_bf = (bf16_t*)carve((size_t)NV * ND * 2);
  bf16_t* Wprj_bf = (bf16_t*)carve((size_t)ND * TWOE * 2);
  bf16_t* enc_bf  = (bf16_t*)carve((size_t)NS * NB * TWOE * 2);
  float*  S_att   = (float*) carve((size_t)NS * NB * ND * 4);   // [s*B+b, ND]
  float*  gates   = (float*) carve((size_t)NB * NG * 4);
  float*  h       = (float*) carve((size_t)NB * ND * 4);
  float*  c       = (float*) carve((size_t)NB * ND * 4);
  bf16_t* h_bf    = (bf16_t*)carve((size_t)NB * ND * 2);
  bf16_t* x_bf    = (bf16_t*)carve((size_t)NB * (NE + ND) * 2); // [emb | av]
  bf16_t* hc_bf   = (bf16_t*)carve((size_t)NB * (ND + TWOE) * 2);
  (void)ws_size; (void)in_sizes; (void)n_in; (void)out_size;

  // ---- one-time: casts + precomputed attention scores ----
  launch_cast(stream, W_ih0, Wih0_bf, (long long)NG * (NE + ND));
  launch_cast(stream, W_hh0, Whh0_bf, (long long)NG * ND);
  launch_cast(stream, W_ih1, Wih1_bf, (long long)NG * ND);
  launch_cast(stream, W_hh1, Whh1_bf, (long long)NG * ND);
  launch_cast(stream, W_comb, Wcmb_bf, (long long)ND * (TWOE + ND));
  launch_cast(stream, W_out, Wout_bf, (long long)NV * ND);
  launch_cast(stream, W_proj, Wprj_bf, (long long)ND * TWOE);
  launch_cast(stream, src_enc, enc_bf, (long long)NS * NB * TWOE);

  // S_att[(s*B+b), :] = enc[(s*B+b), :] @ W_proj^T  (M=2048: direct-global A)
  launch_gemm(stream, enc_bf, TWOE, TWOE, Wprj_bf,
              nullptr, 0, 0, nullptr, nullptr, nullptr,
              S_att, ND, nullptr, 0, NS * NB, ND, 0, /*stage_a=*/0);

  init_state_kernel<<<(NB * ND + 255) / 256, 256, 0, stream>>>(
      h0, c0, h, c, h_bf, x_bf);

  bf16_t* av_bf = x_bf + NE;   // prev-context slot inside x, ld = NE+ND

  // ---- sequential decode ----
  for (int t = 0; t < NT; ++t) {
    gather_emb_kernel<<<(NB * NE + 255) / 256, 256, 0, stream>>>(
        emb, tgt, x_bf, t);

    // cell 0: gates = x @ W_ih0^T + h @ W_hh0^T + b_ih0 + b_hh0
    launch_gemm(stream, x_bf, NE + ND, NE + ND, Wih0_bf,
                h_bf, ND, ND, Whh0_bf, b_ih0, b_hh0,
                gates, NG, nullptr, 0, NB, NG, 0, /*stage_a=*/1);
    lstm_update_kernel<<<(NB * ND + 255) / 256, 256, 0, stream>>>(
        gates, h, c, h_bf);

    // cell 1: gates = h @ W_ih1^T + h @ W_hh1^T + b_ih1 + b_hh1
    launch_gemm(stream, h_bf, ND, ND, Wih1_bf,
                h_bf, ND, ND, Whh1_bf, b_ih1, b_hh1,
                gates, NG, nullptr, 0, NB, NG, 0, /*stage_a=*/1);
    lstm_update_kernel<<<(NB * ND + 255) / 256, 256, 0, stream>>>(
        gates, h, c, h_bf);

    // attention -> hc_bf = bf16([h, context])
    attention_kernel<<<NB, NS, 0, stream>>>(S_att, src_enc, h, hc_bf);

    // av = tanh(hc @ W_combine^T); write bf16 av straight into x's context
    // slot (also the A operand of the logits GEMM)
    launch_gemm(stream, hc_bf, ND + TWOE, ND + TWOE, Wcmb_bf,
                nullptr, 0, 0, nullptr, nullptr, nullptr,
                nullptr, 0, av_bf, NE + ND, NB, ND, 1, /*stage_a=*/1);

    // scores[t] = av @ W_out^T  (M=32, N=32000, K=1024)
    launch_gemm(stream, av_bf, NE + ND, ND, Wout_bf,
                nullptr, 0, 0, nullptr, nullptr, nullptr,
                out + (size_t)t * NB * NV, NV, nullptr, 0, NB, NV, 0,
                /*stage_a=*/1);
  }
}
